// SpikeNN_20212116095556
// MI455X (gfx1250) — matrix operand order repaired and checked by reading, not compiled, not run on hardware
//
#include <hip/hip_runtime.h>

// ---------------------------------------------------------------------------
// LIF spiking net, MI455X (gfx1250).
// Per step: state = state*KOEFF + W @ out_t ; LIF threshold/refractory.
// out_t[:, :512] is always the input spikes, and only neurons [1792,2048)
// are ever observed -> neuron rows [0,512) of W are dead; compute only
// rows [512,2048): GEMM [1536 x 2048] x [2048 x 64] per step, 512 steps.
//
// Precision: spikes are exactly {0,1} in f16. W is split W = Whi + Wlo
// (both f16) giving ~22 mantissa bits; two v_wmma_f32_16x16x32_f16 per
// K-tile accumulate into f32 -> near-fp32 GEMM on the f16 WMMA pipe.
//
// One persistent kernel runs all T steps; steps are separated by an
// agent-scope atomic grid barrier (12 blocks -> co-resident). Spike
// buffers ping-pong so step t can inject x_{t+1} into the next buffer.
// ---------------------------------------------------------------------------

typedef __attribute__((ext_vector_type(16))) _Float16 v16h;
typedef __attribute__((ext_vector_type(8)))  _Float16 v8h;
typedef __attribute__((ext_vector_type(8)))  float    v8f;

#define K_N       2048
#define K_NIN     512
#define K_NOUT    256
#define K_B       64
#define K_T       512
#define K_KOEFF   0.95122942450071400910f   /* exp(-1/20) */

#define K_THREADS 256                        /* 8 wave32 per block            */
#define K_BLOCKS  12                         /* 96 waves = (2048-512)/16 tiles*/
#define K_MBASE   512                        /* first live neuron row         */

// ---------------------------------------------------------------------------
// Agent-scope grid barrier (12 blocks, all co-resident on MI455X).
// Release: __threadfence flushes this CU's dirty lines to device scope.
// Acquire: fence invalidates L0 so other WGPs' spike writes are visible.
// cnt must start at 0 (init kernel); gen may start at any value.
// ---------------------------------------------------------------------------
__device__ __forceinline__ void grid_barrier(unsigned* cnt, unsigned* gen) {
  __threadfence();
  __syncthreads();
  if (threadIdx.x == 0) {
    unsigned g = __hip_atomic_load(gen, __ATOMIC_RELAXED, __HIP_MEMORY_SCOPE_AGENT);
    unsigned a = __hip_atomic_fetch_add(cnt, 1u, __ATOMIC_ACQ_REL, __HIP_MEMORY_SCOPE_AGENT);
    if (a == (unsigned)(K_BLOCKS - 1)) {
      __hip_atomic_store(cnt, 0u, __ATOMIC_RELAXED, __HIP_MEMORY_SCOPE_AGENT);
      __hip_atomic_store(gen, g + 1u, __ATOMIC_RELEASE, __HIP_MEMORY_SCOPE_AGENT);
    } else {
      while (__hip_atomic_load(gen, __ATOMIC_ACQUIRE, __HIP_MEMORY_SCOPE_AGENT) == g)
        __builtin_amdgcn_s_sleep(1);
    }
  }
  __syncthreads();
  __builtin_amdgcn_fence(__ATOMIC_ACQUIRE, "agent");
}

// ---------------------------------------------------------------------------
// W -> (hi, lo) f16 split.  W row-major [N][N].
// ---------------------------------------------------------------------------
__global__ void wsplit_kernel(const float* __restrict__ W,
                              _Float16* __restrict__ Wh,
                              _Float16* __restrict__ Wl) {
  size_t i = (size_t)blockIdx.x * blockDim.x + threadIdx.x;
  if (i < (size_t)K_N * K_N) {
    float w = W[i];
    _Float16 h = (_Float16)w;
    Wh[i] = h;
    Wl[i] = (_Float16)(w - (float)h);
  }
}

// ---------------------------------------------------------------------------
// Zero state/refr, fill spike buffer 0 with x_0 (k<512) / 0, reset barrier.
// spk layout: [batch][neuron] f16 (rows = batch, contiguous K per row).
// ---------------------------------------------------------------------------
__global__ void init_kernel(const float* __restrict__ inp,
                            float* __restrict__ state,
                            float* __restrict__ refr,
                            _Float16* __restrict__ spk0,
                            unsigned* __restrict__ bar) {
  int i = blockIdx.x * blockDim.x + threadIdx.x;    // 0 .. B*N-1
  if (i < K_B * K_N) {
    state[i] = 0.0f;
    refr[i]  = 0.0f;
    int b = i >> 11, k = i & (K_N - 1);
    _Float16 v = (_Float16)0.0f;
    if (k < K_NIN) v = (_Float16)inp[((size_t)b * K_NIN + k) * K_T + 0];
    spk0[i] = v;
  }
  if (i == 0) bar[0] = 0u;   // arrival count; generation word may be anything
}

// ---------------------------------------------------------------------------
// Persistent LIF kernel: 96 waves, one 16-neuron M-tile x 64 batches each.
// ---------------------------------------------------------------------------
__global__ __launch_bounds__(K_THREADS, 1)
void lif_persistent_kernel(const float* __restrict__ inp,
                           const _Float16* __restrict__ Wh,
                           const _Float16* __restrict__ Wl,
                           float* __restrict__ state,
                           float* __restrict__ refr,
                           _Float16* __restrict__ spkA,
                           _Float16* __restrict__ spkB,
                           float* __restrict__ out,
                           unsigned* __restrict__ bar) {
  const int lane = threadIdx.x & 31;
  const int wid  = blockIdx.x * (K_THREADS / 32) + (threadIdx.x >> 5); // 0..95
  const int m0   = K_MBASE + (wid << 4);          // tile rows m0..m0+15
  // A-frag (16x32 f16): lane holds row M=lane&15;
  // h[0..15] = K {0..15}+off (contiguous, matches B), off = lane<16?0:16.
  const int  rowA = m0 + (lane & 15);
  const int  offA = (lane < 16) ? 0 : 16;
  const int  offB = (lane < 16) ? 0 : 16;         // B-frag: 16 contiguous K
  const _Float16* wrh = Wh + (size_t)rowA * K_N;
  const _Float16* wrl = Wl + (size_t)rowA * K_N;
  // Input-injection mapping: 8-element chunks of [batch][j<512].
  const int tid = blockIdx.x * K_THREADS + threadIdx.x;

  for (int t = 0; t < K_T; ++t) {
    const _Float16* cur = (t & 1) ? spkB : spkA;
    _Float16*       nxt = (t & 1) ? spkA : spkB;

    // ---- GEMM: acc[j] (16x16 f32) = W[tile] @ spikes[:, 16j..16j+15] ----
    v8f acc[4] = {};
#pragma unroll 4
    for (int kt = 0; kt < K_N / 32; ++kt) {
      const int ka = kt * 32 + offA;
      v8h ah0 = *(const v8h*)(wrh + ka);
      v8h ah1 = *(const v8h*)(wrh + ka + 8);
      v8h al0 = *(const v8h*)(wrl + ka);
      v8h al1 = *(const v8h*)(wrl + ka + 8);
      v16h ah = __builtin_shufflevector(ah0, ah1, 0,1,2,3,4,5,6,7,8,9,10,11,12,13,14,15);
      v16h al = __builtin_shufflevector(al0, al1, 0,1,2,3,4,5,6,7,8,9,10,11,12,13,14,15);
      const _Float16* bp = cur + (size_t)(lane & 15) * K_N + (kt * 32 + offB);
#pragma unroll
      for (int j = 0; j < 4; ++j) {
        v16h bf = *(const v16h*)(bp + (size_t)j * 16 * K_N);
        acc[j] = __builtin_amdgcn_wmma_f32_16x16x32_f16(false, ah, false, bf,
                                                        (short)0, acc[j], false, false);
        acc[j] = __builtin_amdgcn_wmma_f32_16x16x32_f16(false, al, false, bf,
                                                        (short)0, acc[j], false, false);
      }
    }

    // ---- LIF update.  C/D frag: VGPR r -> M = m0 + r + (lane>=16 ? 8 : 0),
    //                              N (batch col) = 16j + (lane&15). ----
    const int mB = m0 + ((lane < 16) ? 0 : 8);
#pragma unroll
    for (int j = 0; j < 4; ++j) {
      const int b = (j << 4) + (lane & 15);
      float* sp = state + (size_t)b * K_N + mB;
      float* rp = refr  + (size_t)b * K_N + mB;
      float4 s01 = *(const float4*)sp;
      float4 s23 = *(const float4*)(sp + 4);
      float4 r01 = *(const float4*)rp;
      float4 r23 = *(const float4*)(rp + 4);
      float st[8] = { s01.x, s01.y, s01.z, s01.w, s23.x, s23.y, s23.z, s23.w };
      float rf[8] = { r01.x, r01.y, r01.z, r01.w, r23.x, r23.y, r23.z, r23.w };
      v8h   spk;
      float fv[8];
#pragma unroll
      for (int r = 0; r < 8; ++r) {
        float s = st[r] * K_KOEFF + acc[j][r];
        if (rf[r] > 0.0f) s = 0.0f;          // refractory clamp (uses old refr)
        const bool fire = (s >= 1.0f);
        float rr = rf[r] - 1.0f;
        if (fire) rr = 2.0f;
        if (s < -1.0f) s = -1.0f;
        st[r] = s; rf[r] = rr;
        fv[r] = fire ? 1.0f : 0.0f;
        spk[r] = fire ? (_Float16)1.0f : (_Float16)0.0f;
      }
      *(float4*)sp       = make_float4(st[0], st[1], st[2], st[3]);
      *(float4*)(sp + 4) = make_float4(st[4], st[5], st[6], st[7]);
      *(float4*)rp       = make_float4(rf[0], rf[1], rf[2], rf[3]);
      *(float4*)(rp + 4) = make_float4(rf[4], rf[5], rf[6], rf[7]);
      // Recurrent spikes for step t+1 (all tile rows are >= 512).
      *(v8h*)(nxt + (size_t)b * K_N + mB) = spk;
      // Observed outputs: neurons [1792,2048) -> out[b][m-1792][t].
      if (m0 >= K_N - K_NOUT) {
        float* op = out + ((size_t)b * K_NOUT + (mB - (K_N - K_NOUT))) * K_T + t;
#pragma unroll
        for (int r = 0; r < 8; ++r) op[(size_t)r * K_T] = fv[r];
      }
    }

    // ---- Inject x_{t+1} into next buffer, K slots [0,512). ----
    if (t + 1 < K_T) {
      for (int c = tid; c < (K_B * K_NIN) / 8; c += K_BLOCKS * K_THREADS) {
        const int b  = c >> 6;           // 64 chunks of 8 per batch row
        const int j0 = (c & 63) << 3;
        v8h xi;
#pragma unroll
        for (int q = 0; q < 8; ++q)
          xi[q] = (_Float16)inp[((size_t)b * K_NIN + (j0 + q)) * K_T + (t + 1)];
        *(v8h*)(nxt + (size_t)b * K_N + j0) = xi;
      }
    }

    grid_barrier(bar, bar + 1);
  }
}

// ---------------------------------------------------------------------------
// Launch: W split + init + one persistent kernel (3 graph nodes total).
// Workspace layout (~17.5 MB):
//   [0,8)      barrier {cnt, gen}
//   +256       state  f32 [64][2048]   512 KB
//   +           refr   f32 [64][2048]   512 KB
//   +           spkA   f16 [64][2048]   256 KB
//   +           spkB   f16 [64][2048]   256 KB
//   +           Whi    f16 [2048][2048]   8 MB
//   +           Wlo    f16 [2048][2048]   8 MB
// ---------------------------------------------------------------------------
extern "C" void kernel_launch(void* const* d_in, const int* in_sizes, int n_in,
                              void* d_out, int out_size, void* d_ws, size_t ws_size,
                              hipStream_t stream) {
  (void)in_sizes; (void)n_in; (void)out_size; (void)ws_size;
  const float* inp = (const float*)d_in[0];   // [B, N_IN, T] f32 (0/1)
  const float* W   = (const float*)d_in[1];   // [N, N] f32
  float*       out = (float*)d_out;           // [B, N_OUT, T] f32

  char* ws = (char*)d_ws;
  unsigned* bar   = (unsigned*)ws;
  float*    state = (float*)(ws + 256);
  float*    refr  = state + (size_t)K_B * K_N;
  _Float16* spkA  = (_Float16*)(refr + (size_t)K_B * K_N);
  _Float16* spkB  = spkA + (size_t)K_B * K_N;
  _Float16* Wh    = spkB + (size_t)K_B * K_N;
  _Float16* Wl    = Wh + (size_t)K_N * K_N;

  wsplit_kernel<<<(K_N * K_N) / 256, 256, 0, stream>>>(W, Wh, Wl);
  init_kernel<<<(K_B * K_N) / 256, 256, 0, stream>>>(inp, state, refr, spkA, bar);
  lif_persistent_kernel<<<K_BLOCKS, K_THREADS, 0, stream>>>(
      inp, Wh, Wl, state, refr, spkA, spkB, out, bar);
}
